// MolformerForSequenceClassification_24824910971289
// MI455X (gfx1250) — matrix operand order repaired and checked by reading, not compiled, not run on hardware
//
#include <hip/hip_runtime.h>
#include <hip/hip_bf16.h>

// ---------------------------------------------------------------------------
// Molformer forward pass for MI455X (gfx1250, wave32, WMMA bf16)
// Big GEMMs (QKV / Wo / FFN / classifier) use v_wmma_f32_16x16x32_bf16 with
// LDS tiles stored directly in WMMA fragment order (vector ds loads).
// ---------------------------------------------------------------------------

typedef __attribute__((ext_vector_type(16))) __bf16 v16bf;
typedef __attribute__((ext_vector_type(8)))  __bf16 v8bf;
typedef __attribute__((ext_vector_type(8)))  float  v8f;

#define Bc   64
#define Sc   512
#define Dc   768
#define Hc   12
#define NLc  12
#define Mc   32
#define DHc  64
#define BSc  (Bc * Sc)     // 32768 tokens

__device__ __forceinline__ float gelu_exact(float x) {
    return 0.5f * x * (1.0f + erff(x * 0.70710678118654752f));
}

// --------------------------- cast fp32 -> bf16 -----------------------------
__global__ void cast_f32_bf16_kernel(const float* __restrict__ in,
                                     __bf16* __restrict__ out, int n) {
    int i = blockIdx.x * blockDim.x + threadIdx.x;
    if (i < n) out[i] = (__bf16)in[i];
}

// ------------------------------ embedding ----------------------------------
__global__ void embed_kernel(const int* __restrict__ ids,
                             const float* __restrict__ emb,
                             float* __restrict__ x) {
    int token = blockIdx.x;                 // 0..BS-1
    int id = ids[token];
    const float* src = emb + (size_t)id * Dc;
    float* dst = x + (size_t)token * Dc;
    for (int d = threadIdx.x; d < Dc; d += blockDim.x) dst[d] = src[d];
}

// ------------------------- WMMA bf16 GEMM kernel ---------------------------
// C[M,N] = act( A[M,K] @ B[K,N] + bias[N] + residual[M,N] )
// Workgroup: 256 threads = 8 waves; tile 64x64; K stepped by 32.
// Waves: wr = wave&3 (M subtile), wc = wave>>2 (32-col half); each wave
// computes two 16x16 output tiles sharing one A fragment.
// LDS holds tiles in WMMA *fragment* order so fragment reads are b128s.
#define GTM 64
#define GTN 64
#define GTK 32
#define ALD 24   // per-lane A fragment stride in bf16 (16 used + 8 pad, 48B)
#define BLD 40   // per-column B stride in bf16 (32 used + 8 pad, 80B)

__global__ __launch_bounds__(256)
void gemm_bf16_kernel(const __bf16* __restrict__ A,
                      const __bf16* __restrict__ Bw,
                      const float* __restrict__ bias,
                      const float* __restrict__ residual,
                      float* __restrict__ C,
                      int M, int N, int K, int act /*0=none,1=gelu*/) {
    // A fragments: [4 M-subtiles][32 lanes][ALD]
    __shared__ __align__(16) __bf16 AsF[4 * 32 * ALD];
    // B fragments: [64 columns][BLD] (k runs contiguously within a column)
    __shared__ __align__(16) __bf16 BsF[GTN * BLD];

    const int tid  = threadIdx.x;
    const int lane = tid & 31;
    const int wave = tid >> 5;
    const int wr   = wave & 3;     // M subtile 0..3
    const int wc   = wave >> 2;    // N half 0..1
    const int rowBase = blockIdx.x * GTM;
    const int colBase = blockIdx.y * GTN;

    // ---- A staging: 64x32 bf16, 8 per thread, one b128 store in frag order
    const int arow  = tid >> 2;            // 0..63 (M within tile)
    const int acol  = (tid & 3) * 8;       // 0,8,16,24 (K within tile)
    // fragment mapping: k 0-7 -> lane m, j0..7 ; k 8-15 -> lane m+16, j0..7
    //                   k16-23 -> lane m, j8..15; k24-31 -> lane m+16, j8..15
    const int atile = arow >> 4;
    const int alane = (arow & 15) + ((acol & 8) ? 16 : 0);
    const int ajb   = (acol & 16) ? 8 : 0;
    __bf16* aDst = &AsF[(atile * 32 + alane) * ALD + ajb];

    // ---- B staging: 32x64 bf16, 8 per thread, scatter b16 in frag order
    const int brow = tid >> 3;             // 0..31 (K within tile)
    const int bcol = (tid & 7) * 8;        // 0..56 (N within tile)

    // ---- fragment read mapping
    const int mrow   = lane & 15;
    const int laneHi = lane >> 4;          // 0/1
    const __bf16* aFrag = &AsF[(wr * 32 + lane) * ALD];
    const __bf16* bFrag0 = &BsF[(wc * 32 + mrow) * BLD + laneHi * 8];
    const __bf16* bFrag1 = &BsF[(wc * 32 + 16 + mrow) * BLD + laneHi * 8];

    const __bf16* aG = &A[(size_t)(rowBase + arow) * K + acol];
    const __bf16* bG = &Bw[(size_t)brow * N + colBase + bcol];

    v8f acc0 = {}, acc1 = {};

#pragma unroll 2
    for (int k0 = 0; k0 < K; k0 += GTK) {
        const uint4 av = *reinterpret_cast<const uint4*>(aG + k0);
        const uint4 bv = *reinterpret_cast<const uint4*>(bG + (size_t)k0 * N);
        *reinterpret_cast<uint4*>(aDst) = av;
        const __bf16* be = reinterpret_cast<const __bf16*>(&bv);
#pragma unroll
        for (int i = 0; i < 8; ++i)
            BsF[(bcol + i) * BLD + brow] = be[i];

        if (k0 + GTK < K) {   // prefetch next tiles (global_prefetch_b8)
            __builtin_prefetch(aG + k0 + GTK, 0, 1);
            __builtin_prefetch(bG + (size_t)(k0 + GTK) * N, 0, 1);
        }
        __syncthreads();

        // fragment reads: aligned 16B vector LDS loads
        const v8bf a_lo = *reinterpret_cast<const v8bf*>(aFrag);
        const v8bf a_hi = *reinterpret_cast<const v8bf*>(aFrag + 8);
        const v8bf b0lo = *reinterpret_cast<const v8bf*>(bFrag0);
        const v8bf b0hi = *reinterpret_cast<const v8bf*>(bFrag0 + 16);
        const v8bf b1lo = *reinterpret_cast<const v8bf*>(bFrag1);
        const v8bf b1hi = *reinterpret_cast<const v8bf*>(bFrag1 + 16);
        const v16bf af = __builtin_shufflevector(a_lo, a_hi,
            0,1,2,3,4,5,6,7,8,9,10,11,12,13,14,15);
        const v16bf bf0 = __builtin_shufflevector(b0lo, b0hi,
            0,1,2,3,4,5,6,7,8,9,10,11,12,13,14,15);
        const v16bf bf1 = __builtin_shufflevector(b1lo, b1hi,
            0,1,2,3,4,5,6,7,8,9,10,11,12,13,14,15);

        acc0 = __builtin_amdgcn_wmma_f32_16x16x32_bf16(
            false, af, false, bf0, (short)0, acc0, false, false);
        acc1 = __builtin_amdgcn_wmma_f32_16x16x32_bf16(
            false, af, false, bf1, (short)0, acc1, false, false);
        __syncthreads();
    }

    // epilogue: bias + residual + activation, fp32 out
    const int crow0 = rowBase + wr * 16 + (laneHi ? 8 : 0);
    const int ccol0 = colBase + wc * 32 + mrow;
    const float b0 = bias ? bias[ccol0] : 0.0f;
    const float b1 = bias ? bias[ccol0 + 16] : 0.0f;
#pragma unroll
    for (int r = 0; r < 8; ++r) {
        const int crow = crow0 + r;
        float v0 = acc0[r] + b0;
        float v1 = acc1[r] + b1;
        if (residual) {
            v0 += residual[(size_t)crow * N + ccol0];
            v1 += residual[(size_t)crow * N + ccol0 + 16];
        }
        if (act == 1) { v0 = gelu_exact(v0); v1 = gelu_exact(v1); }
        C[(size_t)crow * N + ccol0]      = v0;
        C[(size_t)crow * N + ccol0 + 16] = v1;
    }
}

// -------------------------------- RoPE -------------------------------------
__global__ void rope_kernel(float* __restrict__ q, int n) {
    int idx = blockIdx.x * blockDim.x + threadIdx.x;   // over BS*H*(DH/2)
    if (idx >= n) return;
    const int half = DHc / 2;
    int i   = idx % half;
    int rem = idx / half;
    int h   = rem % Hc;
    int tok = rem / Hc;
    int s   = tok % Sc;
    float invf = powf(10000.0f, -(float)(2 * i) / (float)DHc);
    float ang  = (float)s * invf;
    float c = cosf(ang), sn = sinf(ang);
    float* base = q + (size_t)tok * Dc + h * DHc;
    float x1 = base[i], x2 = base[i + half];
    base[i]        = x1 * c - x2 * sn;
    base[i + half] = x2 * c + x1 * sn;
}

// ----------------------- random features: relu(q @ omega) ------------------
__global__ void feature_kernel(const float* __restrict__ q,
                               const float* __restrict__ om,   // DH x M
                               const float* __restrict__ mask, // B*S
                               float* __restrict__ pq, int n, int applyMask) {
    int idx = blockIdx.x * blockDim.x + threadIdx.x;   // over BS*H*M
    if (idx >= n) return;
    int mcol = idx % Mc;
    int rem  = idx / Mc;
    int h    = rem % Hc;
    int bs   = rem / Hc;
    const float* qp = q + (size_t)bs * Dc + h * DHc;
    float s = 0.0f;
#pragma unroll 8
    for (int d = 0; d < DHc; ++d) s += qp[d] * om[d * Mc + mcol];
    s = fmaxf(s, 0.0f);
    if (applyMask) s *= mask[bs];
    pq[idx] = s;
}

// -------------------- kv[b,h,m,d] = sum_s pk * v ---------------------------
__global__ void kv_kernel(const float* __restrict__ pk,
                          const float* __restrict__ v,
                          float* __restrict__ kv, int n) {
    int idx = blockIdx.x * blockDim.x + threadIdx.x;   // over B*H*M*DH
    if (idx >= n) return;
    int d   = idx % DHc;
    int rem = idx / DHc;
    int m   = rem % Mc;
    int h   = (rem / Mc) % Hc;
    int b   = rem / (Mc * Hc);
    float s = 0.0f;
    for (int t = 0; t < Sc; ++t) {
        int bs = b * Sc + t;
        s += pk[((size_t)bs * Hc + h) * Mc + m] * v[(size_t)bs * Dc + h * DHc + d];
    }
    kv[idx] = s;
}

// -------------------- pksum[b,h,m] = sum_s pk ------------------------------
__global__ void pksum_kernel(const float* __restrict__ pk,
                             float* __restrict__ pks, int n) {
    int idx = blockIdx.x * blockDim.x + threadIdx.x;   // over B*H*M
    if (idx >= n) return;
    int m = idx % Mc;
    int h = (idx / Mc) % Hc;
    int b = idx / (Mc * Hc);
    float s = 0.0f;
    for (int t = 0; t < Sc; ++t)
        s += pk[((size_t)(b * Sc + t) * Hc + h) * Mc + m];
    pks[idx] = s;
}

// ------ att[b,s,h,d] = (pq . kv[:,d]) / (pq . pksum + 1e-6) ----------------
__global__ void att_kernel(const float* __restrict__ pq,
                           const float* __restrict__ kv,
                           const float* __restrict__ pks,
                           float* __restrict__ att, int n) {
    int idx = blockIdx.x * blockDim.x + threadIdx.x;   // over BS*H*DH
    if (idx >= n) return;
    int d   = idx % DHc;
    int rem = idx / DHc;
    int h   = rem % Hc;
    int bs  = rem / Hc;
    int b   = bs / Sc;
    const float* pqp  = pq  + ((size_t)bs * Hc + h) * Mc;
    const float* pksp = pks + ((size_t)b * Hc + h) * Mc;
    const float* kvp  = kv  + ((size_t)(b * Hc + h)) * Mc * DHc + d;
    float den = 1e-6f, num = 0.0f;
#pragma unroll 8
    for (int m = 0; m < Mc; ++m) {
        float p = pqp[m];
        den += p * pksp[m];
        num += p * kvp[(size_t)m * DHc];
    }
    att[(size_t)bs * Dc + h * DHc + d] = num / den;
}

// ------------------------------ LayerNorm ----------------------------------
__global__ __launch_bounds__(256)
void ln_kernel(const float* __restrict__ x, const float* __restrict__ g,
               const float* __restrict__ bta, float* __restrict__ out) {
    __shared__ float red[256];
    int row = blockIdx.x;
    const float* xp = x + (size_t)row * Dc;
    float s = 0.0f;
    for (int d = threadIdx.x; d < Dc; d += 256) s += xp[d];
    red[threadIdx.x] = s; __syncthreads();
    for (int w = 128; w > 0; w >>= 1) {
        if (threadIdx.x < w) red[threadIdx.x] += red[threadIdx.x + w];
        __syncthreads();
    }
    float mean = red[0] / (float)Dc;
    __syncthreads();
    float v = 0.0f;
    for (int d = threadIdx.x; d < Dc; d += 256) {
        float t = xp[d] - mean; v += t * t;
    }
    red[threadIdx.x] = v; __syncthreads();
    for (int w = 128; w > 0; w >>= 1) {
        if (threadIdx.x < w) red[threadIdx.x] += red[threadIdx.x + w];
        __syncthreads();
    }
    float rstd = rsqrtf(red[0] / (float)Dc + 1e-5f);
    float* op = out + (size_t)row * Dc;
    for (int d = threadIdx.x; d < Dc; d += 256)
        op[d] = (xp[d] - mean) * rstd * g[d] + bta[d];
}

// ------------------------------ mean pool ----------------------------------
__global__ void pool_kernel(const float* __restrict__ enc,
                            const float* __restrict__ mask,
                            float* __restrict__ pooled, int n) {
    int idx = blockIdx.x * blockDim.x + threadIdx.x;   // over B*D
    if (idx >= n) return;
    int b = idx / Dc, d = idx % Dc;
    float s = 0.0f, ms = 0.0f;
    for (int t = 0; t < Sc; ++t) {
        float m = mask[b * Sc + t];
        s  += enc[(size_t)(b * Sc + t) * Dc + d] * m;
        ms += m;
    }
    pooled[idx] = s / fmaxf(ms, 1e-9f);
}

// ------------------------------ classifier head ----------------------------
__global__ void logits_kernel(const float* __restrict__ h,
                              const float* __restrict__ Wc2,
                              const float* __restrict__ bc2,
                              float* __restrict__ out) {
    int b = blockIdx.x * blockDim.x + threadIdx.x;
    if (b >= Bc) return;
    float s = bc2[0];
    for (int d = 0; d < Dc; ++d) s += h[(size_t)b * Dc + d] * Wc2[d];
    out[b] = s;
}

// ===========================================================================
extern "C" void kernel_launch(void* const* d_in, const int* in_sizes, int n_in,
                              void* d_out, int out_size, void* d_ws, size_t ws_size,
                              hipStream_t stream) {
    const int*   ids   = (const int*)  d_in[0];
    const float* mask  = (const float*)d_in[1];
    const float* emb   = (const float*)d_in[2];
    const float* Wq    = (const float*)d_in[3];
    const float* bq    = (const float*)d_in[4];
    const float* Wk    = (const float*)d_in[5];
    const float* bk    = (const float*)d_in[6];
    const float* Wv    = (const float*)d_in[7];
    const float* bv    = (const float*)d_in[8];
    const float* Wo    = (const float*)d_in[9];
    const float* bo    = (const float*)d_in[10];
    const float* omega = (const float*)d_in[11];
    const float* ln1g  = (const float*)d_in[12];
    const float* ln1b  = (const float*)d_in[13];
    const float* Wf1   = (const float*)d_in[14];
    const float* bf1   = (const float*)d_in[15];
    const float* Wf2   = (const float*)d_in[16];
    const float* bf2   = (const float*)d_in[17];
    const float* ln2g  = (const float*)d_in[18];
    const float* ln2b  = (const float*)d_in[19];
    const float* lnfg  = (const float*)d_in[20];
    const float* lnfb  = (const float*)d_in[21];
    const float* Wc1   = (const float*)d_in[22];
    const float* bc1   = (const float*)d_in[23];
    const float* Wc2   = (const float*)d_in[24];
    const float* bc2   = (const float*)d_in[25];
    float* outp = (float*)d_out;

    // ---- workspace layout (256B aligned slices) ----
    char* p = (char*)d_ws;
    auto take = [&](size_t bytes) -> char* {
        char* r = p; p += (bytes + 255) & ~(size_t)255; return r;
    };
    const size_t nX  = (size_t)BSc * Dc;          // token activations
    const size_t nPQ = (size_t)BSc * Hc * Mc;
    const size_t nKV = (size_t)Bc * Hc * Mc * DHc;
    const size_t nPS = (size_t)Bc * Hc * Mc;
    const size_t nW  = (size_t)NLc * Dc * Dc;

    float*  xf   = (float*) take(nX * 4);
    __bf16* xb   = (__bf16*)take(nX * 2);
    float*  qf   = (float*) take(nX * 4);
    float*  kf   = (float*) take(nX * 4);
    float*  vf   = (float*) take(nX * 4);      // also holds att output
    float*  pqb  = (float*) take(nPQ * 4);
    float*  pkb  = (float*) take(nPQ * 4);
    float*  kvb  = (float*) take(nKV * 4);
    float*  pks  = (float*) take(nPS * 4);
    float*  t1   = (float*) take(nX * 4);      // FFN intermediate
    __bf16* wqb  = (__bf16*)take(nW * 2);
    __bf16* wkb  = (__bf16*)take(nW * 2);
    __bf16* wvb  = (__bf16*)take(nW * 2);
    __bf16* wob  = (__bf16*)take(nW * 2);
    __bf16* wf1b = (__bf16*)take(nW * 2);
    __bf16* wf2b = (__bf16*)take(nW * 2);
    __bf16* wc1b = (__bf16*)take((size_t)Dc * Dc * 2);
    float*  pooled  = (float*) take((size_t)Bc * Dc * 4);
    __bf16* pooledb = (__bf16*)take((size_t)Bc * Dc * 2);
    float*  hbuf    = (float*) take((size_t)Bc * Dc * 4);

    auto grid1 = [](size_t n) { return dim3((unsigned)((n + 255) / 256)); };
    const dim3 blk(256);

    // ---- cast all GEMM weights to bf16 once per call ----
    cast_f32_bf16_kernel<<<grid1(nW), blk, 0, stream>>>(Wq,  wqb,  (int)nW);
    cast_f32_bf16_kernel<<<grid1(nW), blk, 0, stream>>>(Wk,  wkb,  (int)nW);
    cast_f32_bf16_kernel<<<grid1(nW), blk, 0, stream>>>(Wv,  wvb,  (int)nW);
    cast_f32_bf16_kernel<<<grid1(nW), blk, 0, stream>>>(Wo,  wob,  (int)nW);
    cast_f32_bf16_kernel<<<grid1(nW), blk, 0, stream>>>(Wf1, wf1b, (int)nW);
    cast_f32_bf16_kernel<<<grid1(nW), blk, 0, stream>>>(Wf2, wf2b, (int)nW);
    cast_f32_bf16_kernel<<<grid1((size_t)Dc * Dc), blk, 0, stream>>>(Wc1, wc1b, Dc * Dc);

    // ---- embedding gather ----
    embed_kernel<<<dim3(BSc), blk, 0, stream>>>(ids, emb, xf);

    const dim3 ggrid(BSc / GTM, Dc / GTN);   // (512, 12)
    const int nRope = BSc * Hc * (DHc / 2);
    const int nFeat = BSc * Hc * Mc;
    const int nAtt  = BSc * Hc * DHc;

    for (int l = 0; l < NLc; ++l) {
        const size_t wOff = (size_t)l * Dc * Dc;
        const size_t bOff = (size_t)l * Dc;

        // --- QKV projections (WMMA bf16) ---
        cast_f32_bf16_kernel<<<grid1(nX), blk, 0, stream>>>(xf, xb, (int)nX);
        gemm_bf16_kernel<<<ggrid, blk, 0, stream>>>(xb, wqb + wOff, bq + bOff, nullptr, qf, BSc, Dc, Dc, 0);
        gemm_bf16_kernel<<<ggrid, blk, 0, stream>>>(xb, wkb + wOff, bk + bOff, nullptr, kf, BSc, Dc, Dc, 0);
        gemm_bf16_kernel<<<ggrid, blk, 0, stream>>>(xb, wvb + wOff, bv + bOff, nullptr, vf, BSc, Dc, Dc, 0);

        // --- RoPE ---
        rope_kernel<<<grid1(nRope), blk, 0, stream>>>(qf, nRope);
        rope_kernel<<<grid1(nRope), blk, 0, stream>>>(kf, nRope);

        // --- random features ---
        const float* om = omega + (size_t)l * DHc * Mc;
        feature_kernel<<<grid1(nFeat), blk, 0, stream>>>(qf, om, mask, pqb, nFeat, 0);
        feature_kernel<<<grid1(nFeat), blk, 0, stream>>>(kf, om, mask, pkb, nFeat, 1);

        // --- linear attention ---
        kv_kernel   <<<grid1(nKV), blk, 0, stream>>>(pkb, vf, kvb, (int)nKV);
        pksum_kernel<<<grid1(nPS), blk, 0, stream>>>(pkb, pks, (int)nPS);
        att_kernel  <<<grid1(nAtt), blk, 0, stream>>>(pqb, kvb, pks, vf, nAtt);

        // --- output projection + residual ---
        cast_f32_bf16_kernel<<<grid1(nX), blk, 0, stream>>>(vf, xb, (int)nX);
        gemm_bf16_kernel<<<ggrid, blk, 0, stream>>>(xb, wob + wOff, bo + bOff, xf, qf, BSc, Dc, Dc, 0);
        ln_kernel<<<dim3(BSc), blk, 0, stream>>>(qf, ln1g + bOff, ln1b + bOff, xf);

        // --- FFN: gelu(x@Wf1+b) @ Wf2 + b + residual ---
        cast_f32_bf16_kernel<<<grid1(nX), blk, 0, stream>>>(xf, xb, (int)nX);
        gemm_bf16_kernel<<<ggrid, blk, 0, stream>>>(xb, wf1b + wOff, bf1 + bOff, nullptr, t1, BSc, Dc, Dc, 1);
        cast_f32_bf16_kernel<<<grid1(nX), blk, 0, stream>>>(t1, xb, (int)nX);
        gemm_bf16_kernel<<<ggrid, blk, 0, stream>>>(xb, wf2b + wOff, bf2 + bOff, xf, qf, BSc, Dc, Dc, 0);
        ln_kernel<<<dim3(BSc), blk, 0, stream>>>(qf, ln2g + bOff, ln2b + bOff, xf);
    }

    // ---- final LN + masked mean pool + classifier ----
    ln_kernel<<<dim3(BSc), blk, 0, stream>>>(xf, lnfg, lnfb, qf);
    pool_kernel<<<grid1((size_t)Bc * Dc), blk, 0, stream>>>(qf, mask, pooled, Bc * Dc);
    cast_f32_bf16_kernel<<<grid1((size_t)Bc * Dc), blk, 0, stream>>>(pooled, pooledb, Bc * Dc);
    gemm_bf16_kernel<<<dim3(Bc / GTM, Dc / GTN), blk, 0, stream>>>(
        pooledb, wc1b, bc1, nullptr, hbuf, Bc, Dc, Dc, 1);
    logits_kernel<<<dim3(1), dim3(64), 0, stream>>>(hbuf, Wc2, bc2, outp);
}
